// WeightOnlyInt8Linear_51393578664547
// MI455X (gfx1250) — compile-verified, matrix operand order checked by reading, not run on hardware
//
#include <hip/hip_runtime.h>

// WeightOnlyInt8 grouped-dequant linear for MI455X (gfx1250, wave32).
// out[16,28672] = x[16,8192] @ (int8 W * per-(row,group128) scale)^T
// Memory-bound: 0.94 GB of weights @ 23.3 TB/s => ~40us floor.
// Math path: V_WMMA_F32_16X16X32_F16, f32 accumulate, 4-way K-split + LDS reduce.

typedef __attribute__((ext_vector_type(16))) _Float16 v16h;
typedef __attribute__((ext_vector_type(8)))  float    v8f;
typedef __attribute__((ext_vector_type(4)))  int      i32x4;
typedef __attribute__((ext_vector_type(4)))  float    f32x4;

#define TOKENS 16
#define IN_F   8192
#define OUT_F  28672
#define GS     128
#define NG     (IN_F / GS)      // 64 groups
#define KCHUNK (IN_F / 4)       // 2048 K per wave (4 waves per block)

__global__ __launch_bounds__(128) void WeightOnlyInt8Linear_wmma_kernel(
    const float* __restrict__ x,      // [16, 8192] f32
    const int*   __restrict__ w,      // [28672, 8192] int8-valued int32
    const float* __restrict__ scales, // [28672, 64] f32
    float*       __restrict__ out)    // [16, 28672] f32
{
    __shared__ float red[4][256];     // 4 waves x 16x16 f32 partial tiles

    const int tid   = threadIdx.x;
    const int lane  = tid & 31;
    const int wv    = tid >> 5;            // wave 0..3 -> K chunk
    const int obase = blockIdx.x << 4;     // 16 output channels per block

    // A-fragment: lane holds out-channel row M = lane&15.
    // B-fragment: lane holds token column N = lane&15 (layout mirrors A).
    // Per ISA 16-bit 16x32 layout, a lane's 16 K values are two contiguous
    // runs of 8: [kb+c, kb+c+8) and [kb+c+16, kb+c+24), c = 8*(lane>=16).
    const int  row   = obase + (lane & 15);
    const int  tok   = lane & 15;
    const int  chalf = (lane >> 4) << 3;   // 0 or 8
    const long kbase = (long)wv * KCHUNK;

    const int*   __restrict__ wrow = w + (long)row * IN_F;
    const float* __restrict__ xrow = x + (long)tok * IN_F;
    const float* __restrict__ srow = scales + (long)row * NG;

    v8f acc = {};

    for (int g = 0; g < KCHUNK / GS; ++g) {        // 16 groups of 128 K
        const long  kg = kbase + (long)g * GS;
        const float s  = srow[kg >> 7];            // lane-uniform over A frag
#pragma unroll
        for (int kk = 0; kk < 4; ++kk) {           // 4 x K=32 WMMA per group
            const long kb = kg + kk * 32 + chalf;

            // Weights: stream once, non-temporal (don't pollute L2 vs x).
            i32x4 wa0 = __builtin_nontemporal_load((const i32x4*)(wrow + kb));
            i32x4 wa1 = __builtin_nontemporal_load((const i32x4*)(wrow + kb + 4));
            i32x4 wb0 = __builtin_nontemporal_load((const i32x4*)(wrow + kb + 16));
            i32x4 wb1 = __builtin_nontemporal_load((const i32x4*)(wrow + kb + 20));
            // x: tiny, stays L2/WGP$-resident.
            f32x4 xa0 = *(const f32x4*)(xrow + kb);
            f32x4 xa1 = *(const f32x4*)(xrow + kb + 4);
            f32x4 xb0 = *(const f32x4*)(xrow + kb + 16);
            f32x4 xb1 = *(const f32x4*)(xrow + kb + 20);

            v16h A, B;
#pragma unroll
            for (int i = 0; i < 4; ++i) {
                A[i]      = (_Float16)((float)wa0[i] * s);  // int8 exact in f16
                A[4 + i]  = (_Float16)((float)wa1[i] * s);
                A[8 + i]  = (_Float16)((float)wb0[i] * s);
                A[12 + i] = (_Float16)((float)wb1[i] * s);
                B[i]      = (_Float16)xa0[i];
                B[4 + i]  = (_Float16)xa1[i];
                B[8 + i]  = (_Float16)xb0[i];
                B[12 + i] = (_Float16)xb1[i];
            }

            // D = A(16x32) x B(32x16) + C, f32 16x16 accumulator.
            acc = __builtin_amdgcn_wmma_f32_16x16x32_f16(
                false, A, false, B, (short)0, acc, false, false);
        }
    }

    // C/D layout: VGPR v, lane l -> M = v + 8*(l>=16), N = l&15.
    const int mhi = (lane >> 4) << 3;
#pragma unroll
    for (int v = 0; v < 8; ++v)
        red[wv][(v + mhi) * 16 + tok] = acc[v];

    __syncthreads();

    // Reduce 4 K-partials; out is [tokens, OUT_F].
#pragma unroll
    for (int j = tid; j < 256; j += 128) {
        float sum = red[0][j] + red[1][j] + red[2][j] + red[3][j];
        const int M = j >> 4, N = j & 15;
        out[(long)N * OUT_F + obase + M] = sum;
    }
}

extern "C" void kernel_launch(void* const* d_in, const int* in_sizes, int n_in,
                              void* d_out, int out_size, void* d_ws, size_t ws_size,
                              hipStream_t stream) {
    const float* x      = (const float*)d_in[0];
    const int*   weight = (const int*)d_in[1];
    const float* scales = (const float*)d_in[2];
    float*       out    = (float*)d_out;

    dim3 grid(OUT_F / 16);   // 1792 tiles of 16 output channels
    dim3 block(128);         // 4 waves: 4-way K split
    WeightOnlyInt8Linear_wmma_kernel<<<grid, block, 0, stream>>>(x, weight, scales, out);
}
